// dozer_attention_14929306321692
// MI455X (gfx1250) — compile-verified, matrix operand order checked by reading, not run on hardware
//
#include <hip/hip_runtime.h>

// Dozer sparse attention, CDNA5 (gfx1250, wave32, WMMA + async-to-LDS).
//
// Reference math collapses: V[b,n,t,h,d] = (sum_l softmax(...)[t,l]) * v[b,n,h,t,d],
// i.e. out is v permuted times a row factor ~= 1.0.  We faithfully compute the
// factor via WMMA-based Q*K^T + masked softmax row-sums, then scale v.
// Memory-bound: ~134 MB traffic -> ~6 us at 23.3 TB/s; compute is kept under
// that bound by exploiting the mask structure (1 live strided row per lane).

typedef __attribute__((ext_vector_type(16))) _Float16 v16h;
typedef __attribute__((ext_vector_type(8)))  _Float16 v8h;
typedef __attribute__((ext_vector_type(8)))  float    v8f;
typedef int v4i __attribute__((vector_size(4 * sizeof(int))));

#define T_DIM 256
#define D_DIM 32
#define N_DIM 64
#define H_DIM 8
// log2(e) * 32^-0.5 : folded into the f16 A fragment so that the native
// v_exp_f32 (which computes 2^x) needs no per-element multiply.
#define ASCALE 0.2550348663f

// Raw v_exp_f32 (2^x): our arguments are masked scores in ~[-10, 10] or 0,
// so the generic exp2's denormal guard (cmp/add/exp/ldexp) is dead weight.
#if __has_builtin(__builtin_amdgcn_exp2f)
#define EXP2F(x) __builtin_amdgcn_exp2f(x)
#else
#define EXP2F(x) __builtin_exp2f(x)
#endif

#if defined(__AMDGCN__) && \
    __has_builtin(__builtin_amdgcn_global_load_async_to_lds_b128) && \
    __has_builtin(__builtin_amdgcn_s_wait_asynccnt)
#define USE_ASYNC 1
typedef __attribute__((address_space(1))) v4i* gptr_b128;  // global int4*
typedef __attribute__((address_space(3))) v4i* lptr_b128;  // LDS int4*
#else
#define USE_ASYNC 0
#endif

__global__ __launch_bounds__(256) void dozer_attn_kernel(
    const float* __restrict__ q,
    const float* __restrict__ k,
    const float* __restrict__ v,
    float* __restrict__ out)
{
    // 256x32 K tile as f16, shared by all 8 waves (16 KB).
    __shared__ __align__(16) _Float16 Kh[T_DIM * D_DIM];
#if USE_ASYNC
    // Raw f32 staging buffer filled by GLOBAL_LOAD_ASYNC_TO_LDS (32 KB).
    __shared__ __align__(16) float Kstage[T_DIM * D_DIM];
#endif

    const int bid  = blockIdx.x;            // ((b*64 + n)*8 + h)
    const int h    = bid & 7;
    const int n    = (bid >> 3) & 63;
    const int b    = bid >> 9;
    const int base = bid * (T_DIM * D_DIM); // flat base into q/k/v (b,n,h,t,d)

    const int tid  = threadIdx.x;
    const int lane = tid & 31;
    const int wave = tid >> 5;
    const int g    = lane >> 4;   // half-wave group: owns C rows 8g..8g+7
    const int sub  = lane & 15;   // column / row-in-tile index
    const int k0   = g * 8;       // K-dim split for 16-bit A/B fragments

    // ---- stage K into LDS as f16 (one 32-element row per thread) ----
#if USE_ASYNC
    {
        const float* src = k + base + tid * D_DIM;
        float*       stg = &Kstage[tid * D_DIM];
        #pragma unroll
        for (int j = 0; j < D_DIM; j += 4) {
            __builtin_amdgcn_global_load_async_to_lds_b128(
                (gptr_b128)(src + j),
                (lptr_b128)(stg + j),
                /*offset=*/0, /*cpol=*/0);
        }
        __builtin_amdgcn_s_wait_asynccnt(0);   // own row staged -> convert it
        _Float16* dst = &Kh[tid * D_DIM];
        #pragma unroll
        for (int j = 0; j < D_DIM; j += 4) {
            float4 f = *(const float4*)(stg + j);
            dst[j + 0] = (_Float16)f.x;
            dst[j + 1] = (_Float16)f.y;
            dst[j + 2] = (_Float16)f.z;
            dst[j + 3] = (_Float16)f.w;
        }
    }
#else
    {
        const float* kr = k + base + tid * D_DIM;
        _Float16* dst = &Kh[tid * D_DIM];
        #pragma unroll
        for (int j = 0; j < D_DIM; j += 4) {
            float4 f = *(const float4*)(kr + j);
            dst[j + 0] = (_Float16)f.x;
            dst[j + 1] = (_Float16)f.y;
            dst[j + 2] = (_Float16)f.z;
            dst[j + 3] = (_Float16)f.w;
        }
    }
#endif
    __syncthreads();

    // Strided-diagonal mask (t-l)%8==0 is k-tile invariant: with tloc0%8==0
    // and kbase%8==0 it reduces to (r-sub)%8==0.  Per lane EXACTLY ONE of the
    // 8 C-rows is "live" (r == sub&7); the rest contribute exp2(0)=1 on tiles
    // outside the local band.
    bool strided[8];
    #pragma unroll
    for (int r = 0; r < 8; ++r) strided[r] = (((sub - r) & 7) == 0);

    // ---- each wave handles 2 query tiles of 16 rows ----
    #pragma unroll
    for (int qi = 0; qi < 2; ++qi) {
        const int qt    = wave * 2 + qi;
        const int qbase = qt * 16;
        const int qtU   = __builtin_amdgcn_readfirstlane(qt); // wave-uniform

        // A fragment: 16x32 f16, ISA layout (lane%16 = row M; lanes<16 hold
        // K 0..7 & 16..23, lanes>=16 hold K 8..15 & 24..31).  Pre-scaled by
        // log2(e)*d^-0.5 so the k-loop feeds v_exp_f32 (2^x) directly.
        const float* qrow = q + base + (qbase + sub) * D_DIM;
        v16h a;
        #pragma unroll
        for (int i = 0; i < 8; ++i) {
            a[i]     = (_Float16)(qrow[k0 + i]      * ASCALE);
            a[i + 8] = (_Float16)(qrow[k0 + 16 + i] * ASCALE);
        }

        float p[8];          // per-row masked exp2 sums (near tiles + merge)
        #pragma unroll
        for (int r = 0; r < 8; ++r) p[r] = 0.0f;
        float e_live = 0.0f; // far-tile exp2 sum for this lane's live row

        const int tloc0 = qbase + g * 8;  // C-matrix row of c[0] for this lane

        #pragma unroll 4
        for (int kt = 0; kt < 16; ++kt) {
            const int kbase = kt * 16;
            // B fragment: key row (kbase+sub), same K split as A (B = K^T).
            const _Float16* krow = &Kh[(kbase + sub) * D_DIM];
            v8h lo = *(const v8h*)(krow + k0);
            v8h hi = *(const v8h*)(krow + k0 + 16);
            v16h bf = __builtin_shufflevector(lo, hi,
                        0, 1, 2, 3, 4, 5, 6, 7, 8, 9, 10, 11, 12, 13, 14, 15);

            v8f c = {};
            c = __builtin_amdgcn_wmma_f32_16x16x32_f16(
                    /*neg_a=*/false, a, /*neg_b=*/false, bf,
                    /*c_mod=*/(short)0, c, /*reuse_a=*/false, /*reuse_b=*/false);

            // Wave-uniform near/far split: the local band |t-l|<=3 can only
            // hit k-tiles adjacent to this q-tile.
            if ((unsigned)(kt - qtU + 1) <= 2u) {
                // near tile: full mask (band | strided), 8 exps.
                const int dqs = tloc0 - kbase - sub;
                #pragma unroll
                for (int r = 0; r < 8; ++r) {
                    const bool band = (unsigned)(dqs + r + 3) <= 6u;
                    const bool m    = strided[r] | band;
                    const float x   = m ? c[r] : 0.0f;
                    p[r] += EXP2F(x);
                }
            } else {
                // far tile: only the live row has a nonzero score; the other
                // 7 rows contribute exp2(0)=1 each -> added analytically below.
                float sel = 0.0f;
                #pragma unroll
                for (int r = 0; r < 8; ++r) sel = strided[r] ? c[r] : sel;
                e_live += EXP2F(sel);
            }
        }

        // Merge far-tile contributions: farcnt ones for non-live rows,
        // the accumulated exp2 sum for the live row.
        const float farcnt = (qtU == 0 || qtU == 15) ? 14.0f : 13.0f;
        #pragma unroll
        for (int r = 0; r < 8; ++r)
            p[r] += strided[r] ? e_live : farcnt;

        // Row reductions across the 16 lanes that hold each row's columns
        // (xor masks 1,2,4,8 never cross the 16-lane halves of a wave32).
        #pragma unroll
        for (int r = 0; r < 8; ++r) {
            float s = p[r];
            #pragma unroll
            for (int m = 1; m < 16; m <<= 1) s += __shfl_xor(s, m, 32);
            float fp = p[r] / s;          // per-lane partial of sum_l A[t,l]
            #pragma unroll
            for (int m = 1; m < 16; m <<= 1) fp += __shfl_xor(fp, m, 32);
            p[r] = fp;                    // ~= 1.0: faithful softmax row-sum
        }

        // out[b, t, n, h*32 + dd] = factor(t) * v[b,n,h,t,dd]
        // 16 lanes x float2 per row -> 128B coalesced stores per half-wave.
        #pragma unroll
        for (int r = 0; r < 8; ++r) {
            const int t = tloc0 + r;
            const float2 vv = *(const float2*)(v + base + t * D_DIM + 2 * sub);
            float2 o;
            o.x = p[r] * vv.x;
            o.y = p[r] * vv.y;
            float* op = out
                + ((b * T_DIM + t) * N_DIM + n) * (H_DIM * D_DIM)
                + h * D_DIM + 2 * sub;
            *(float2*)op = o;
        }
    }
}

extern "C" void kernel_launch(void* const* d_in, const int* in_sizes, int n_in,
                              void* d_out, int out_size, void* d_ws, size_t ws_size,
                              hipStream_t stream) {
    (void)in_sizes; (void)n_in; (void)out_size; (void)d_ws; (void)ws_size;
    const float* q = (const float*)d_in[0];
    const float* k = (const float*)d_in[1];
    const float* v = (const float*)d_in[2];
    float* out = (float*)d_out;
    // One block per (b, n, h): 2*64*8 = 1024 blocks, 256 threads = 8 waves.
    dozer_attn_kernel<<<dim3(1024), dim3(256), 0, stream>>>(q, k, v, out);
}
